// MultiHeadAttention_48954037240110
// MI455X (gfx1250) — matrix operand order repaired and checked by reading, not compiled, not run on hardware
//
#include <hip/hip_runtime.h>
#include <hip/hip_bf16.h>
#include <math.h>

// Problem constants (match reference)
#define NUM_HEADS    16
#define NUM_KV_HEADS 4
#define HEAD_DIM     128
#define DIM          2048
#define BB           2
#define SS           2048
#define QDIM   (NUM_HEADS * HEAD_DIM)    // 2048
#define KVDIM  (NUM_KV_HEADS * HEAD_DIM) // 512

typedef __attribute__((ext_vector_type(16))) _Float16 v16h;
typedef __attribute__((ext_vector_type(8)))  _Float16 v8h;
typedef __attribute__((ext_vector_type(8)))  float    v8f;

// A-fragment loader (contiguous k order for WMMA 16x16x32 f16; B operands use the same): for a lane with half-select
// hs, elements 0..15 are K = base+hs*16.. (contiguous)
// Two aligned 16B vector loads concatenated into one SSA v16h so the backend
// lands them directly in the even-aligned 8-VGPR WMMA source tuple.
__device__ __forceinline__ v16h load_frag16(const _Float16* p) {
  v8h lo = *(const v8h*)p;
  v8h hi = *(const v8h*)(p + 8);
  return __builtin_shufflevector(lo, hi, 0, 1, 2, 3, 4, 5, 6, 7,
                                 8, 9, 10, 11, 12, 13, 14, 15);
}

// ---- wave32 cross-lane reductions over the 16-lane half-groups -------------
__device__ __forceinline__ float red_max16(float x) {
#pragma unroll
  for (int m = 1; m < 16; m <<= 1) x = fmaxf(x, __shfl_xor(x, m, 32));
  return x;
}
__device__ __forceinline__ float red_sum16(float x) {
#pragma unroll
  for (int m = 1; m < 16; m <<= 1) x += __shfl_xor(x, m, 32);
  return x;
}

// ---- register-blocked f16 WMMA GEMM: C[MxN] = A[MxK] * Bt[NxK]^T ----------
// Each wave computes a 64x32 output block (4 M-tiles x 2 N-tiles of 16x16):
// 8 WMMAs per 6KB of operand traffic (~21 FLOP/byte). 4 waves/block over N.
__global__ __launch_bounds__(128)
void gemm_f16_tn(const _Float16* __restrict__ A,
                 const _Float16* __restrict__ Bt,
                 float* __restrict__ C,
                 int M, int N, int K)
{
  const int lane = threadIdx.x & 31;
  const int wave = threadIdx.x >> 5;
  const int n0 = (blockIdx.x * 4 + wave) * 32;  // 2 N-tiles per wave
  const int m0 = blockIdx.y * 64;               // 4 M-tiles per wave
  if (n0 >= N || m0 >= M) return;               // wave-uniform
  const int nl = lane & 15;                     // row (A) / col (B,C) in tile
  const int hs = lane >> 4;                     // lane-half select

  const _Float16* arow[4];
#pragma unroll
  for (int mi = 0; mi < 4; ++mi)
    arow[mi] = A + (size_t)(m0 + mi * 16 + nl) * K + hs * 16;
  const _Float16* brow[2];
#pragma unroll
  for (int ni = 0; ni < 2; ++ni)
    brow[ni] = Bt + (size_t)(n0 + ni * 16 + nl) * K + hs * 16;

  v8f acc[4][2] = {};
  for (int kk = 0; kk < K; kk += 32) {
    v16h a[4];
#pragma unroll
    for (int mi = 0; mi < 4; ++mi)
      a[mi] = load_frag16(arow[mi] + kk);       // A-layout fragment
    v16h b[2];
#pragma unroll
    for (int ni = 0; ni < 2; ++ni)
      b[ni] = *(const v16h*)(brow[ni] + kk);    // 16 contiguous K values

    if (kk + 128 < K) {                         // global_prefetch_b8
      __builtin_prefetch(arow[0] + kk + 128, 0, 3);
      __builtin_prefetch(brow[0] + kk + 128, 0, 3);
    }

#pragma unroll
    for (int mi = 0; mi < 4; ++mi)
#pragma unroll
      for (int ni = 0; ni < 2; ++ni)
        acc[mi][ni] = __builtin_amdgcn_wmma_f32_16x16x32_f16(
            false, a[mi], false, b[ni], (short)0, acc[mi][ni], false, false);
  }

#pragma unroll
  for (int mi = 0; mi < 4; ++mi)
#pragma unroll
    for (int ni = 0; ni < 2; ++ni)
#pragma unroll
      for (int r = 0; r < 8; ++r)
        C[(size_t)(m0 + mi * 16 + hs * 8 + r) * N + n0 + ni * 16 + nl] =
            acc[mi][ni][r];
}

// ---- elementwise conversion kernels ---------------------------------------
__global__ void f32_to_f16(const float* __restrict__ in,
                           _Float16* __restrict__ out, size_t n) {
  size_t id = (size_t)blockIdx.x * blockDim.x + threadIdx.x;
  if (id < n) out[id] = (_Float16)in[id];
}

// W is KxN row-major; Wt becomes NxK row-major (f16)
__global__ void transpose_f32_to_f16(const float* __restrict__ W,
                                     _Float16* __restrict__ Wt, int K, int N) {
  size_t id = (size_t)blockIdx.x * blockDim.x + threadIdx.x;
  if (id >= (size_t)K * N) return;
  int n = (int)(id % N);
  int k = (int)(id / N);
  Wt[(size_t)n * K + k] = (_Float16)W[id];
}

// RoPE on (B,S,nheads,HEAD_DIM) f32, fold in optional score scale, emit f16
__global__ void rope_f32_to_f16(const float* __restrict__ X,
                                const float* __restrict__ fc,
                                const float* __restrict__ fs,
                                _Float16* __restrict__ out,
                                int nheads, float scale)
{
  size_t id = (size_t)blockIdx.x * blockDim.x + threadIdx.x;
  const int HP = HEAD_DIM / 2;
  size_t total = (size_t)BB * SS * nheads * HP;
  if (id >= total) return;
  int    i = (int)(id % HP);
  int    h = (int)((id / HP) % nheads);
  size_t s = (id / ((size_t)HP * nheads)) % SS;
  size_t b = id / ((size_t)HP * nheads * SS);
  int width = nheads * HEAD_DIM;
  size_t base = (b * SS + s) * width + (size_t)h * HEAD_DIM + 2 * i;
  float tr = X[base], ti = X[base + 1];
  float c = fc[s * HP + i], sn = fs[s * HP + i];
  out[base]     = (_Float16)((tr * c - ti * sn) * scale);
  out[base + 1] = (_Float16)((tr * sn + ti * c) * scale);
}

// V: (B,S,KVDIM) f32 -> Vt: (B,KVH,HEAD_DIM,S) f16 so PV B-fragments are
// contiguous over the key dimension.
__global__ void v_transpose(const float* __restrict__ V,
                            _Float16* __restrict__ Vt) {
  size_t id = (size_t)blockIdx.x * blockDim.x + threadIdx.x;
  size_t total = (size_t)BB * SS * KVDIM;
  if (id >= total) return;
  int    c = (int)(id % KVDIM);
  size_t s = (id / KVDIM) % SS;
  size_t b = id / ((size_t)KVDIM * SS);
  int kvh = c / HEAD_DIM, d = c % HEAD_DIM;
  Vt[((b * NUM_KV_HEADS + kvh) * (size_t)HEAD_DIM + d) * SS + s] = (_Float16)V[id];
}

// ---- causal flash attention: one wave per (b, h, 16-query tile) -----------
__global__ __launch_bounds__(128)
void attn_kernel(const _Float16* __restrict__ Qh,  // (B,S,QDIM), pre-scaled
                 const _Float16* __restrict__ Kh,  // (B,S,KVDIM)
                 const _Float16* __restrict__ Vt,  // (B,KVH,HEAD_DIM,S)
                 _Float16* __restrict__ Oh)        // (B,S,QDIM)
{
  __shared__ __align__(32) _Float16 Pb[4][16 * 32];  // per-wave P relayout tile
  const int lane = threadIdx.x & 31;
  const int wave = threadIdx.x >> 5;
  const int nl = lane & 15;
  const int hs = lane >> 4;
  const int qt = blockIdx.x * 4 + wave;
  const int h  = blockIdx.y;
  const int b  = blockIdx.z;
  const int kvh = h / (NUM_HEADS / NUM_KV_HEADS);
  const int q0 = qt * 16;

  const _Float16* qbase = Qh + (size_t)b * SS * QDIM + (size_t)h * HEAD_DIM;
  const _Float16* kbase = Kh + (size_t)b * SS * KVDIM + (size_t)kvh * HEAD_DIM;
  const _Float16* vbase = Vt + ((size_t)(b * NUM_KV_HEADS + kvh)) * HEAD_DIM * SS;

  // Q A-fragments: 4 K-chunks of 32 covering the 128 head dims (stay resident)
  v16h qa[4];
  {
    const _Float16* qrow = qbase + (size_t)(q0 + nl) * QDIM + hs * 16;
#pragma unroll
    for (int c4 = 0; c4 < 4; ++c4) qa[c4] = load_frag16(qrow + c4 * 32);
  }

  v8f o[8] = {};                          // full 16x128 f32 O tile
  float mr[8], lr[8];
#pragma unroll
  for (int r = 0; r < 8; ++r) { mr[r] = -__builtin_inff(); lr[r] = 0.f; }

  const int kend = q0 + 16;               // causal: keys 0 .. q0+15
  for (int j = 0; j < kend; j += 32) {
    if (j + 32 < kend) {                  // prefetch next key block rows
      __builtin_prefetch(kbase + (size_t)(j + 32 + nl) * KVDIM, 0, 3);
      __builtin_prefetch(vbase + (size_t)nl * SS + j + 32, 0, 3);
    }
    // scores for 32 keys: two 16x16 tiles
    v8f s0 = {}, s1 = {};
#pragma unroll
    for (int c4 = 0; c4 < 4; ++c4) {
      v16h b0 = *(const v16h*)(kbase + (size_t)(j + nl) * KVDIM + c4 * 32 + hs * 16);
      v16h b1 = *(const v16h*)(kbase + (size_t)(j + 16 + nl) * KVDIM + c4 * 32 + hs * 16);
      s0 = __builtin_amdgcn_wmma_f32_16x16x32_f16(false, qa[c4], false, b0,
                                                  (short)0, s0, false, false);
      s1 = __builtin_amdgcn_wmma_f32_16x16x32_f16(false, qa[c4], false, b1,
                                                  (short)0, s1, false, false);
    }
    // causal mask + online softmax (scale already folded into Qh)
    const int kA = j + nl, kB = j + 16 + nl;
    float p0[8], p1[8], al[8];
#pragma unroll
    for (int r = 0; r < 8; ++r) {
      const int row = q0 + hs * 8 + r;
      float a0 = (kA <= row) ? s0[r] : -__builtin_inff();
      float a1 = (kB <= row) ? s1[r] : -__builtin_inff();
      float mt = red_max16(fmaxf(a0, a1));
      float mn = fmaxf(mr[r], mt);
      al[r] = __expf(mr[r] - mn);        // exp(-inf) = 0 on first block
      mr[r] = mn;
      p0[r] = __expf(a0 - mn);
      p1[r] = __expf(a1 - mn);
      lr[r] = lr[r] * al[r] + red_sum16(p0[r] + p1[r]);
    }
#pragma unroll
    for (int t = 0; t < 8; ++t)
#pragma unroll
      for (int r = 0; r < 8; ++r) o[t][r] *= al[r];

    // P: C-layout regs -> LDS -> A-layout fragment (wave-private, no barrier)
#pragma unroll
    for (int r = 0; r < 8; ++r) {
      Pb[wave][(hs * 8 + r) * 32 + nl]      = (_Float16)p0[r];
      Pb[wave][(hs * 8 + r) * 32 + 16 + nl] = (_Float16)p1[r];
    }
    v16h pa = load_frag16(&Pb[wave][nl * 32 + hs * 16]);
    // O += P(16x32) * V(32x16) over 8 dim-tiles
#pragma unroll
    for (int t = 0; t < 8; ++t) {
      v16h vb = *(const v16h*)(vbase + (size_t)(t * 16 + nl) * SS + j + hs * 16);
      o[t] = __builtin_amdgcn_wmma_f32_16x16x32_f16(false, pa, false, vb,
                                                    (short)0, o[t], false, false);
    }
  }

  // normalize and emit f16 for the O-projection GEMM
#pragma unroll
  for (int r = 0; r < 8; ++r) {
    const int row = q0 + hs * 8 + r;
    const float inv = 1.f / lr[r];
    _Float16* orow = Oh + (size_t)(b * SS + row) * QDIM + (size_t)h * HEAD_DIM;
#pragma unroll
    for (int t = 0; t < 8; ++t) orow[t * 16 + nl] = (_Float16)(o[t][r] * inv);
  }
}

// ---------------------------------------------------------------------------
extern "C" void kernel_launch(void* const* d_in, const int* in_sizes, int n_in,
                              void* d_out, int out_size, void* d_ws, size_t ws_size,
                              hipStream_t stream)
{
  const float* x  = (const float*)d_in[0];
  const float* fc = (const float*)d_in[1];
  const float* fs = (const float*)d_in[2];
  const float* wq = (const float*)d_in[3];
  const float* wk = (const float*)d_in[4];
  const float* wv = (const float*)d_in[5];
  const float* wo = (const float*)d_in[6];
  float* out = (float*)d_out;

  // Workspace carve-out (~124 MB total), 256B aligned slices
  char* p = (char*)d_ws;
  auto alloc = [&](size_t bytes) -> char* {
    char* r = p; p += (bytes + 255) & ~(size_t)255; return r;
  };
  const size_t M = (size_t)BB * SS;  // 4096 token rows
  _Float16* xh  = (_Float16*)alloc(M * DIM * 2);
  _Float16* wqT = (_Float16*)alloc((size_t)QDIM  * DIM  * 2);
  _Float16* wkT = (_Float16*)alloc((size_t)KVDIM * DIM  * 2);
  _Float16* wvT = (_Float16*)alloc((size_t)KVDIM * DIM  * 2);
  _Float16* woT = (_Float16*)alloc((size_t)DIM   * QDIM * 2);
  float*    Qf  = (float*)alloc(M * QDIM  * 4);
  float*    Kf  = (float*)alloc(M * KVDIM * 4);
  float*    Vf  = (float*)alloc(M * KVDIM * 4);
  _Float16* Qh  = (_Float16*)alloc(M * QDIM  * 2);
  _Float16* Kh  = (_Float16*)alloc(M * KVDIM * 2);
  _Float16* Vt  = (_Float16*)alloc(M * KVDIM * 2);
  _Float16* Ah  = (_Float16*)alloc(M * QDIM  * 2);

  const int T = 256;
  f32_to_f16<<<(unsigned)((M * DIM + T - 1) / T), T, 0, stream>>>(x, xh, M * DIM);
  transpose_f32_to_f16<<<(unsigned)(((size_t)DIM * QDIM  + T - 1) / T), T, 0, stream>>>(wq, wqT, DIM,  QDIM);
  transpose_f32_to_f16<<<(unsigned)(((size_t)DIM * KVDIM + T - 1) / T), T, 0, stream>>>(wk, wkT, DIM,  KVDIM);
  transpose_f32_to_f16<<<(unsigned)(((size_t)DIM * KVDIM + T - 1) / T), T, 0, stream>>>(wv, wvT, DIM,  KVDIM);
  transpose_f32_to_f16<<<(unsigned)(((size_t)QDIM * DIM  + T - 1) / T), T, 0, stream>>>(wo, woT, QDIM, DIM);

  // QKV projections (register-blocked WMMA GEMM; block = 64 rows x 128 cols)
  gemm_f16_tn<<<dim3(QDIM  / 128, (unsigned)(M / 64)), 128, 0, stream>>>(xh, wqT, Qf, (int)M, QDIM,  DIM);
  gemm_f16_tn<<<dim3(KVDIM / 128, (unsigned)(M / 64)), 128, 0, stream>>>(xh, wkT, Kf, (int)M, KVDIM, DIM);
  gemm_f16_tn<<<dim3(KVDIM / 128, (unsigned)(M / 64)), 128, 0, stream>>>(xh, wvT, Vf, (int)M, KVDIM, DIM);

  // RoPE + downcast (fold 1/sqrt(HEAD_DIM) into Q), V transpose for PV GEMM
  const float sc = 0.08838834764831845f;  // 1/sqrt(128)
  rope_f32_to_f16<<<(unsigned)((M * NUM_HEADS    * (HEAD_DIM / 2) + T - 1) / T), T, 0, stream>>>(Qf, fc, fs, Qh, NUM_HEADS, sc);
  rope_f32_to_f16<<<(unsigned)((M * NUM_KV_HEADS * (HEAD_DIM / 2) + T - 1) / T), T, 0, stream>>>(Kf, fc, fs, Kh, NUM_KV_HEADS, 1.0f);
  v_transpose<<<(unsigned)((M * KVDIM + T - 1) / T), T, 0, stream>>>(Vf, Vt);

  // Causal flash attention (WMMA QK^T and PV)
  attn_kernel<<<dim3(SS / 16 / 4, NUM_HEADS, BB), 128, 0, stream>>>(Qh, Kh, Vt, Ah);

  // Output projection (WMMA), f32 straight into d_out
  gemm_f16_tn<<<dim3(DIM / 128, (unsigned)(M / 64)), 128, 0, stream>>>(Ah, woT, out, (int)M, DIM, QDIM);
}
